// PyTorchRNN_15599321219389
// MI455X (gfx1250) — matrix operand order repaired and checked by reading, not compiled, not run on hardware
//
#include <hip/hip_runtime.h>
#include <hip/hip_bf16.h>
#include <math.h>

// Problem sizes (match reference)
#define V_DIM 50257
#define H_DIM 512
#define B_DIM 64
#define S_DIM 1024

#define NBLK1 8                 // persistent workgroups in the recurrence
#define NCOL_WG (H_DIM / NBLK1) // 64 output columns per workgroup
#define K2_NCOL 128             // logits kernel: columns per workgroup

typedef __bf16 bf16;
typedef __attribute__((ext_vector_type(16))) __bf16 v16bf;
typedef __attribute__((ext_vector_type(8)))  float  v8f;

union FragB16 { v16bf v; uint4 q[2]; };

__device__ __forceinline__ unsigned short f32_to_bf16_bits(float f) {
  unsigned u = __builtin_bit_cast(unsigned, f);
  unsigned r = u + 0x7FFFu + ((u >> 16) & 1u);   // round-to-nearest-even
  return (unsigned short)(r >> 16);
}
__device__ __forceinline__ bf16 f2bf(float f) {
  unsigned short s = f32_to_bf16_bits(f);
  return __builtin_bit_cast(bf16, s);
}
__device__ __forceinline__ uint4 pack8_bf16(float4 a, float4 b) {
  uint4 r;
  r.x = (unsigned)f32_to_bf16_bits(a.x) | ((unsigned)f32_to_bf16_bits(a.y) << 16);
  r.y = (unsigned)f32_to_bf16_bits(a.z) | ((unsigned)f32_to_bf16_bits(a.w) << 16);
  r.z = (unsigned)f32_to_bf16_bits(b.x) | ((unsigned)f32_to_bf16_bits(b.y) << 16);
  r.w = (unsigned)f32_to_bf16_bits(b.z) | ((unsigned)f32_to_bf16_bits(b.w) << 16);
  return r;
}

// Branch-free tanh: tanh(x) = 1 - 2/(exp2(2*log2e*x) + 1).
// Monotone saturation at both ends (exp2 -> inf gives 1, -> 0 gives -1).
// One v_exp_f32 + one v_rcp_f32 + 3 FMA-class ops; accuracy way below bf16 ulp.
__device__ __forceinline__ float fast_tanh(float x) {
  float t = __builtin_amdgcn_exp2f(x * 2.8853900817779268f); // 2*log2(e)
  float r = __builtin_amdgcn_rcpf(t + 1.0f);
  return fmaf(-2.0f, r, 1.0f);
}

// ---------------------------------------------------------------------------
// Kernel 1: persistent RNN scan. 8 WGs x 256 threads (8 wave32 each).
// Each WG owns 64 output columns; per step computes
//   h_out[:, cols] = tanh(WihT[x[:,t]][cols] + bias[cols] + h_in @ Whh[cols,:]^T)
// Whh column-slice lives in LDS (64KB, B-fragment layout) for all 1024 steps.
// Device-scope spin barrier between steps.
// ---------------------------------------------------------------------------
__global__ __launch_bounds__(256)
void rnn_scan_kernel(const long long* __restrict__ x,     // [B][S] token ids
                     const float* __restrict__ WihT,      // [V][H]
                     const float* __restrict__ Whh,       // [H][H]
                     const float* __restrict__ b_ih,      // [H]
                     const float* __restrict__ b_hh,      // [H]
                     bf16* __restrict__ h_buf,            // [2][B][H] (buf0 zeroed)
                     unsigned* __restrict__ bar_cnt)      // zeroed
{
  // B-fragment-layout resident Whh slice: [ntile][kstep][lane][16 bf16] = 64KB
  __shared__ bf16 lds_b[4][16][32][16];

  const int tid    = threadIdx.x;
  const int lane   = tid & 31;
  const int wave   = tid >> 5;
  const int wgbase = blockIdx.x * NCOL_WG;

  // Stage Whh slice -> LDS in WMMA B layout (32x16 bf16 tile):
  //   lane&15 = N column in tile; lanes 0-15 hold K=k0+e, lanes 16-31 hold K=k0+16+e
  //   B[k][n] = Whh[n][k]   (out = h @ Whh^T)
  for (int idx = tid; idx < 4 * 16 * 32 * 16; idx += 256) {
    int e  = idx & 15;
    int l  = (idx >> 4) & 31;
    int ks = (idx >> 9) & 15;
    int nt = (idx >> 13);
    int n  = wgbase + nt * 16 + (l & 15);
    int k  = ks * 32 + ((l >> 4) << 4) + e;
    lds_b[nt][ks][l][e] = f2bf(Whh[n * H_DIM + k]);
  }
  __syncthreads();

  // Tile ownership: wave -> (mtile, two consecutive ntiles)
  const int mtile = wave & 3;
  const int nt0   = (wave >> 2) * 2;
  const int nl16  = lane & 15;

  // Per-lane output columns and folded biases
  const int   nc0   = wgbase + (nt0    ) * 16 + nl16;
  const int   nc1   = wgbase + (nt0 + 1) * 16 + nl16;
  const float bias0 = b_ih[nc0] + b_hh[nc0];
  const float bias1 = b_ih[nc1] + b_hh[nc1];

  // A-fragment (16x32 bf16) addressing, contiguous K per half-wave (k = 16h + s), same order as B
  const int arow      = mtile * 16 + nl16;               // M row this lane loads
  const int ahalf     = ((lane >> 4) & 1) * 16;          // K sub-offset per half-wave
  const int mrow_base = mtile * 16 + ((lane >> 4) << 3); // + r -> M of acc VGPR r

  for (int t = 0; t < S_DIM; ++t) {
    const bf16* __restrict__ hin  = h_buf + (size_t)( t & 1      ) * (B_DIM * H_DIM);
    bf16*       __restrict__ hout = h_buf + (size_t)((t & 1) ^ 1 ) * (B_DIM * H_DIM);

    v8f acc0 = {}; v8f acc1 = {};
#pragma unroll 4
    for (int ks = 0; ks < 16; ++ks) {
      FragB16 a;
      const uint4* ap = reinterpret_cast<const uint4*>(hin + arow * H_DIM + ks * 32 + ahalf);
      a.q[0] = ap[0];          // K = k0 .. k0+7
      a.q[1] = ap[1];          // K = k0+8 .. k0+15
      FragB16 b0, b1;
      const uint4* bp0 = reinterpret_cast<const uint4*>(&lds_b[nt0    ][ks][lane][0]);
      const uint4* bp1 = reinterpret_cast<const uint4*>(&lds_b[nt0 + 1][ks][lane][0]);
      b0.q[0] = bp0[0]; b0.q[1] = bp0[1];
      b1.q[0] = bp1[0]; b1.q[1] = bp1[1];
      acc0 = __builtin_amdgcn_wmma_f32_16x16x32_bf16(false, a.v, false, b0.v,
                                                     (short)0, acc0, false, false);
      acc1 = __builtin_amdgcn_wmma_f32_16x16x32_bf16(false, a.v, false, b1.v,
                                                     (short)0, acc1, false, false);
    }

    // Epilogue: fold token embedding + bias, branch-free tanh, emit bf16 h_out.
    // Also prefetch step t+1's embedding rows into the WGP cache: they are the
    // only unpredictable (token-dependent) accesses on the serial chain.
#pragma unroll
    for (int r = 0; r < 8; ++r) {
      int   m   = mrow_base + r;
      int   tok = (int)x[(size_t)m * S_DIM + t];
      float x0  = WihT[(size_t)tok * H_DIM + nc0];
      float x1  = WihT[(size_t)tok * H_DIM + nc1];
      hout[m * H_DIM + nc0] = f2bf(fast_tanh(acc0[r] + x0 + bias0));
      hout[m * H_DIM + nc1] = f2bf(fast_tanh(acc1[r] + x1 + bias1));
      if (t + 1 < S_DIM) {
        int tokn = (int)x[(size_t)m * S_DIM + (t + 1)];
        __builtin_prefetch(WihT + (size_t)tokn * H_DIM + nc0, 0, 1);
      }
    }

    // Device-scope barrier across the 8 persistent WGs
    __builtin_amdgcn_fence(__ATOMIC_RELEASE, "agent");
    __syncthreads();
    if (tid == 0) {
      atomicAdd(bar_cnt, 1u);
      unsigned need = (unsigned)NBLK1 * (unsigned)(t + 1);
      while (__hip_atomic_load(bar_cnt, __ATOMIC_RELAXED, __HIP_MEMORY_SCOPE_AGENT) < need)
        __builtin_amdgcn_s_sleep(1);
    }
    __syncthreads();
    __builtin_amdgcn_fence(__ATOMIC_ACQUIRE, "agent");
  }
  // h_last ends in h_buf[0] (S even)
}

// ---------------------------------------------------------------------------
// Kernel 2: logits = h_last @ Wfc^T + b_fc.  Bandwidth-bound (Wfc = 103MB).
// 256 threads/WG, each WG computes a 64x128 output tile via WMMA bf16.
// ---------------------------------------------------------------------------
__global__ __launch_bounds__(256)
void logits_kernel(const bf16* __restrict__ h_last,   // [B][H] bf16
                   const float* __restrict__ Wfc,     // [V][H]
                   const float* __restrict__ b_fc,    // [V]
                   float* __restrict__ logits)        // [B][V]
{
  const int tid   = threadIdx.x;
  const int lane  = tid & 31;
  const int wave  = tid >> 5;
  const int base  = blockIdx.x * K2_NCOL;
  const int mtile = wave & 3;
  const int ngrp  = wave >> 2;                 // 0..1 -> 4 ntiles each
  const int nl16  = lane & 15;
  const int khalf = ((lane >> 4) & 1) * 16;    // B-fragment K sub-offset (f32 elems)
  const int arow  = mtile * 16 + nl16;
  const int ahalf = ((lane >> 4) & 1) * 16;

  v8f acc[4] = {{}, {}, {}, {}};

  for (int ks = 0; ks < 16; ++ks) {
    FragB16 a;
    const uint4* ap = reinterpret_cast<const uint4*>(h_last + arow * H_DIM + ks * 32 + ahalf);
    a.q[0] = ap[0];
    a.q[1] = ap[1];
#pragma unroll
    for (int j = 0; j < 4; ++j) {
      int nt = ngrp * 4 + j;
      int n  = base + nt * 16 + nl16;
      FragB16 b;
      if (n < V_DIM) {
        const float4* wp = reinterpret_cast<const float4*>(Wfc + (size_t)n * H_DIM + ks * 32 + khalf);
        float4 w0 = wp[0], w1 = wp[1], w2 = wp[2], w3 = wp[3];
        b.q[0] = pack8_bf16(w0, w1);
        b.q[1] = pack8_bf16(w2, w3);
        if (ks < 15)  // stream-ahead hint -> global_prefetch_b8
          __builtin_prefetch(Wfc + (size_t)n * H_DIM + (ks + 1) * 32 + khalf, 0, 1);
      } else {
        b.q[0] = make_uint4(0, 0, 0, 0);
        b.q[1] = make_uint4(0, 0, 0, 0);
      }
      acc[j] = __builtin_amdgcn_wmma_f32_16x16x32_bf16(false, a.v, false, b.v,
                                                       (short)0, acc[j], false, false);
    }
  }

  const int mrow_base = mtile * 16 + ((lane >> 4) << 3);
#pragma unroll
  for (int j = 0; j < 4; ++j) {
    int n = base + (ngrp * 4 + j) * 16 + nl16;
    if (n >= V_DIM) continue;
    float bias = b_fc[n];
#pragma unroll
    for (int r = 0; r < 8; ++r)
      logits[(size_t)(mrow_base + r) * V_DIM + n] = acc[j][r] + bias;
  }
}

// ---------------------------------------------------------------------------
// Kernel 3: row softmax over V. One WG (256 threads) per batch row.
// ---------------------------------------------------------------------------
__global__ __launch_bounds__(256)
void softmax_kernel(float* __restrict__ logits,   // [B][V] (scratch, overwritten)
                    float* __restrict__ out)      // [B][V]
{
  __shared__ float red[256];
  const int b   = blockIdx.x;
  const int tid = threadIdx.x;
  float* row  = logits + (size_t)b * V_DIM;
  float* orow = out    + (size_t)b * V_DIM;

  float m = -INFINITY;
  for (int i = tid; i < V_DIM; i += 256) m = fmaxf(m, row[i]);
  red[tid] = m; __syncthreads();
  for (int s = 128; s > 0; s >>= 1) {
    if (tid < s) red[tid] = fmaxf(red[tid], red[tid + s]);
    __syncthreads();
  }
  m = red[0];
  __syncthreads();

  float sum = 0.f;
  for (int i = tid; i < V_DIM; i += 256) {
    float e = expf(row[i] - m);
    orow[i] = e;
    sum += e;
  }
  red[tid] = sum; __syncthreads();
  for (int s = 128; s > 0; s >>= 1) {
    if (tid < s) red[tid] += red[tid + s];
    __syncthreads();
  }
  float inv = 1.0f / red[0];
  for (int i = tid; i < V_DIM; i += 256) orow[i] *= inv;
}

// ---------------------------------------------------------------------------
extern "C" void kernel_launch(void* const* d_in, const int* in_sizes, int n_in,
                              void* d_out, int out_size, void* d_ws, size_t ws_size,
                              hipStream_t stream)
{
  (void)in_sizes; (void)n_in; (void)out_size; (void)ws_size;

  const long long* x    = (const long long*)d_in[0];
  const float*     WihT = (const float*)d_in[1];
  const float*     Whh  = (const float*)d_in[2];
  const float*     b_ih = (const float*)d_in[3];
  const float*     b_hh = (const float*)d_in[4];
  const float*     Wfc  = (const float*)d_in[5];
  const float*     b_fc = (const float*)d_in[6];
  float*           out  = (float*)d_out;

  // Workspace layout: [logits f32 B*V][h_buf bf16 2*B*H][barrier counter]
  char*  ws      = (char*)d_ws;
  float* logits  = (float*)ws;
  size_t off     = (size_t)B_DIM * V_DIM * sizeof(float);   // 12.9 MB, 16B aligned
  bf16*  h_buf   = (bf16*)(ws + off);
  size_t hbytes  = 2ull * B_DIM * H_DIM * sizeof(bf16);      // 128 KB
  unsigned* bar  = (unsigned*)(ws + off + hbytes);

  // Zero h0 + barrier state every call (deterministic across graph replays)
  hipMemsetAsync(ws + off, 0, hbytes + 4 * sizeof(unsigned), stream);

  rnn_scan_kernel<<<NBLK1, 256, 0, stream>>>(x, WihT, Whh, b_ih, b_hh, h_buf, bar);
  logits_kernel<<<(V_DIM + K2_NCOL - 1) / K2_NCOL, 256, 0, stream>>>(h_buf, Wfc, b_fc, logits);
  softmax_kernel<<<B_DIM, 256, 0, stream>>>(logits, out);
}
